// SoftmaxAttention_82094004896543
// MI455X (gfx1250) — matrix operand order repaired and checked by reading, not compiled, not run on hardware
//
#include <hip/hip_runtime.h>
#include <stdint.h>

// ---------------------------------------------------------------------------
// Problem constants (from reference): B=2, L=2048, D=1024, H=16, dk=64
// ---------------------------------------------------------------------------
#define BATCH   2
#define SEQ     2048
#define DMODEL  1024
#define NHEAD   16
#define HDIM    64
#define NQKV    (3 * DMODEL)       // 3072
#define MROWS   (BATCH * SEQ)      // 4096

typedef __bf16 bf16_t;
typedef __attribute__((ext_vector_type(16))) __bf16 v16bf;
typedef __attribute__((ext_vector_type(2)))  __bf16 v2bf;
typedef __attribute__((ext_vector_type(8)))  float  v8f;
typedef __attribute__((ext_vector_type(4)))  int    v4i;

#define V8ZERO {0.f,0.f,0.f,0.f,0.f,0.f,0.f,0.f}

// Pin fragments as asm in/out operands: WMMAs consuming them get a data
// dependence on the asm and cannot hoist above it; all fragments are forced
// live in distinct register groups -> one dscnt wait, then a WMMA burst.
#define PIN5(a, b0, b1, b2, b3) \
    asm volatile("" : "+v"(a), "+v"(b0), "+v"(b1), "+v"(b2), "+v"(b3))

// ---------------------------------------------------------------------------
// CDNA5 async global->LDS copy (ASYNCcnt-tracked).  Builtin signature probed:
// param0 = int4 in addrspace(1) (global), param1 = int4 in addrspace(3) (LDS).
// ---------------------------------------------------------------------------
#if defined(__has_builtin)
#  if __has_builtin(__builtin_amdgcn_global_load_async_to_lds_b128)
#    define HAVE_ASYNC 1
#  endif
#  if __has_builtin(__builtin_amdgcn_cvt_pk_bf16_f32)
#    define HAVE_PKBF16 1
#  endif
#endif
#ifndef HAVE_ASYNC
#  define HAVE_ASYNC 0
#endif
#ifndef HAVE_PKBF16
#  define HAVE_PKBF16 0
#endif

#if HAVE_ASYNC
typedef __attribute__((address_space(1))) v4i* gasync_p;
typedef __attribute__((address_space(3))) v4i* lasync_p;
#  if __has_builtin(__builtin_amdgcn_s_wait_asynccnt)
#    define WAIT_ASYNC(n) __builtin_amdgcn_s_wait_asynccnt((unsigned short)(n))
#  else
#    define WAIT_ASYNC(n) asm volatile("s_wait_asynccnt %0" :: "n"(n) : "memory")
#  endif
#else
#  define WAIT_ASYNC(n)
#endif

__device__ __forceinline__ void async_ld_b128(const void* g, void* l) {
#if HAVE_ASYNC
    __builtin_amdgcn_global_load_async_to_lds_b128((gasync_p)g, (lasync_p)l, 0, 0);
#else
    *(uint4*)l = *(const uint4*)g;
#endif
}

// ---------------------------------------------------------------------------
// Helpers
// ---------------------------------------------------------------------------
__device__ __forceinline__ unsigned short f32_to_bf16(float f) {
    unsigned int u = __float_as_uint(f);
    u += 0x7FFFu + ((u >> 16) & 1u);       // round-to-nearest-even
    return (unsigned short)(u >> 16);
}

__device__ __forceinline__ unsigned int pack2_bf16(float x, float y) {
#if HAVE_PKBF16
    union { v2bf v; unsigned int u; } r;
    r.v = __builtin_amdgcn_cvt_pk_bf16_f32(x, y);
    return r.u;
#else
    return (unsigned int)f32_to_bf16(x) | ((unsigned int)f32_to_bf16(y) << 16);
#endif
}

__device__ __forceinline__ v8f wmma_bf16(v16bf a, v16bf b, v8f c) {
    return __builtin_amdgcn_wmma_f32_16x16x32_bf16(
        false, a, false, b, (short)0, c, false, false);
}

// A fragment (16x32 bf16) from row-major [M][K] source, byte stride sB.
// lanes 0-15: M=lane, K=0..7 & 16..23; lanes 16-31: M=lane-16, K=8..15 & 24..31
__device__ __forceinline__ v16bf load_fragA(const unsigned char* base, int sB, int lane) {
    int row = lane & 15, lh = lane >> 4;
    const unsigned char* p = base + row * sB + 16 * lh;
    union { v16bf v; uint4 q[2]; } f;
    f.q[0] = *(const uint4*)(p);
    f.q[1] = *(const uint4*)(p + 32);
    return f.v;
}

// B fragment (32x16 bf16) from [N][K]-major source (K contiguous), stride sB.
// lane: N=lane%16; lanes 0-15 hold K=0..7 & 16..23, lanes 16-31 hold K=8..15 & 24..31
__device__ __forceinline__ v16bf load_fragB(const unsigned char* base, int sB, int lane) {
    int row = lane & 15, lh = lane >> 4;
    const unsigned char* p = base + row * sB + 16 * lh;
    union { v16bf v; uint4 q[2]; } f;
    f.q[0] = *(const uint4*)(p);
    f.q[1] = *(const uint4*)(p + 32);
    return f.v;
}

__device__ __forceinline__ void cvt4_store(unsigned char* dst, float4 f) {
    uint2 d;
    d.x = pack2_bf16(f.x, f.y);
    d.y = pack2_bf16(f.z, f.w);
    *(uint2*)dst = d;
}

// ---------------------------------------------------------------------------
// Kernel 1: QKV projection  qkv = X @ W_qkv^T   (M=4096, N=3072, K=1024)
// Register-staged double-buffered LDS pipeline; scatters into q (pre-scaled),
// k [B,H,L,dk] and vT [B,H,dk,L] bf16 workspaces.
// ---------------------------------------------------------------------------
__global__ __launch_bounds__(256)
void qkv_gemm_wmma(const float* __restrict__ X, const float* __restrict__ W,
                   unsigned short* __restrict__ q_ws,
                   unsigned short* __restrict__ k_ws,
                   unsigned short* __restrict__ vt_ws) {
    __shared__ alignas(16) unsigned char As[2][64 * 80];
    __shared__ alignas(16) unsigned char Bs[2][128 * 80];

    const int tid  = threadIdx.x;
    const int lane = tid & 31;
    const int wave = tid >> 5;
    const int lh   = lane >> 4;
    const int col  = lane & 15;

    const int mB0 = 64  * blockIdx.y;
    const int nB0 = 128 * blockIdx.x;
    const int mW  = 16 * (wave >> 1);
    const int nW  = 64 * (wave & 1);

    v8f acc0 = V8ZERO, acc1 = V8ZERO, acc2 = V8ZERO, acc3 = V8ZERO;

    float4 ar[2], br[4];

    #define LOAD_A(k0)                                                         \
        _Pragma("unroll") for (int i = 0; i < 2; ++i) {                        \
            int idx = tid + 256 * i; int row = idx >> 3, c4 = idx & 7;         \
            ar[i] = *(const float4*)(X + (size_t)(mB0 + row) * DMODEL + (k0) + 4 * c4); }
    #define LOAD_B(k0)                                                         \
        _Pragma("unroll") for (int i = 0; i < 4; ++i) {                        \
            int idx = tid + 256 * i; int row = idx >> 3, c4 = idx & 7;         \
            br[i] = *(const float4*)(W + (size_t)(nB0 + row) * DMODEL + (k0) + 4 * c4); }
    #define STORE_A(buf)                                                       \
        _Pragma("unroll") for (int i = 0; i < 2; ++i) {                        \
            int idx = tid + 256 * i; int row = idx >> 3, c4 = idx & 7;         \
            cvt4_store((buf) + row * 80 + 8 * c4, ar[i]); }
    #define STORE_B(buf)                                                       \
        _Pragma("unroll") for (int i = 0; i < 4; ++i) {                        \
            int idx = tid + 256 * i; int row = idx >> 3, c4 = idx & 7;         \
            cvt4_store((buf) + row * 80 + 8 * c4, br[i]); }

    LOAD_A(0); LOAD_B(0);
    STORE_A(As[0]); STORE_B(Bs[0]);
    __syncthreads();

    for (int kt = 0; kt < DMODEL / 32; ++kt) {
        const int cur = kt & 1;
        const bool pre = (kt + 1 < DMODEL / 32);
        if (pre) { LOAD_A(32 * (kt + 1)); LOAD_B(32 * (kt + 1)); }

        // preload all fragments, pin, then an uninterrupted WMMA burst
        v16bf a  = load_fragA(As[cur] + mW * 80, 80, lane);
        v16bf b0 = load_fragB(Bs[cur] + (nW +  0) * 80, 80, lane);
        v16bf b1 = load_fragB(Bs[cur] + (nW + 16) * 80, 80, lane);
        v16bf b2 = load_fragB(Bs[cur] + (nW + 32) * 80, 80, lane);
        v16bf b3 = load_fragB(Bs[cur] + (nW + 48) * 80, 80, lane);
        PIN5(a, b0, b1, b2, b3);
        acc0 = wmma_bf16(a, b0, acc0);
        acc1 = wmma_bf16(a, b1, acc1);
        acc2 = wmma_bf16(a, b2, acc2);
        acc3 = wmma_bf16(a, b3, acc3);

        if (pre) { STORE_A(As[cur ^ 1]); STORE_B(Bs[cur ^ 1]); }
        __syncthreads();
    }
    #undef LOAD_A
    #undef LOAD_B
    #undef STORE_A
    #undef STORE_B

    // Scatter: n -> (which, head, d); m -> (batch, seqpos)
    const int mB = mB0 + mW;
    const int nB = nB0 + nW;
    v8f accs[4] = { acc0, acc1, acc2, acc3 };
    #pragma unroll
    for (int nt = 0; nt < 4; ++nt) {
        int nTile = nB + 16 * nt;
        int which = nTile >> 10;               // 0=q 1=k 2=v
        int h     = (nTile & 1023) >> 6;
        int d     = (nTile & 63) + col;
        #pragma unroll
        for (int r = 0; r < 8; ++r) {
            int m  = mB + r + 8 * lh;
            int bb = m >> 11;
            int ll = m & 2047;
            size_t bh = (size_t)bb * NHEAD + h;
            float v = accs[nt][r];
            if (which == 0) {
                q_ws[(bh * SEQ + ll) * HDIM + d] = f32_to_bf16(v * 0.125f); // 1/sqrt(64)
            } else if (which == 1) {
                k_ws[(bh * SEQ + ll) * HDIM + d] = f32_to_bf16(v);
            } else {
                vt_ws[(bh * HDIM + d) * SEQ + ll] = f32_to_bf16(v);
            }
        }
    }
}

// ---------------------------------------------------------------------------
// Kernel 2: causal flash attention.
// Grid (L/64, H, B), 128 threads = 4 waves; wave owns 16 q-rows x dk=64 out.
// Double-buffered async global->LDS staging of K / V^T tiles (32 keys),
// online softmax, WMMA for S=QK^T and O+=PV.
// ---------------------------------------------------------------------------
__global__ __launch_bounds__(128)
void flash_attn_wmma(const unsigned short* __restrict__ q_ws,
                     const unsigned short* __restrict__ k_ws,
                     const unsigned short* __restrict__ vt_ws,
                     unsigned short* __restrict__ o_ws) {
    __shared__ alignas(16) unsigned char Ks [2][32 * 144];  // keys x d, padded
    __shared__ alignas(16) unsigned char Vts[2][64 * 80];   // d x keys, padded
    __shared__ alignas(16) unsigned char Ps [4][16 * 80];   // per-wave P transpose

    const int tid  = threadIdx.x;
    const int lane = tid & 31;
    const int wave = tid >> 5;
    const int lh   = lane >> 4;
    const int col  = lane & 15;

    const int b  = blockIdx.z;
    const int h  = blockIdx.y;
    const size_t bh = (size_t)b * NHEAD + h;
    const int q0  = 64 * blockIdx.x;
    const int q0w = q0 + 16 * wave;

    const unsigned short* qp  = q_ws  + bh * SEQ * HDIM;
    const unsigned short* kp  = k_ws  + bh * SEQ * HDIM;
    const unsigned short* vtp = vt_ws + bh * HDIM * SEQ;

    const float NEG_INF = __int_as_float(0xff800000);

    // Q fragments held across the key loop (scale folded in at QKV store)
    v16bf qf0, qf1;
    {
        const unsigned char* qrow = (const unsigned char*)(qp + (size_t)q0w * HDIM);
        qf0 = load_fragA(qrow,      128, lane);   // d = 0..31
        qf1 = load_fragA(qrow + 64, 128, lane);   // d = 32..63
    }

    v8f o0 = V8ZERO, o1 = V8ZERO, o2 = V8ZERO, o3 = V8ZERO;
    float m_run[8], l_run[8];
    #pragma unroll
    for (int r = 0; r < 8; ++r) { m_run[r] = NEG_INF; l_run[r] = 0.f; }

    unsigned char* Psw = Ps[wave];
    const int t_max = (q0 + 63) >> 5;

    // 4 async b128 ops per thread per stage (2 for K, 2 for V^T)
    #define STAGE(buf, k0)                                                         \
        do {                                                                       \
            _Pragma("unroll") for (int i = 0; i < 2; ++i) {                        \
                int idx = tid + 128 * i; int kr = idx >> 3, ch = idx & 7;          \
                async_ld_b128((const unsigned char*)kp + (size_t)(k0) * 128 +      \
                              (size_t)idx * 16, Ks[buf] + kr * 144 + 16 * ch); }   \
            _Pragma("unroll") for (int i = 0; i < 2; ++i) {                        \
                int idx = tid + 128 * i; int dr = idx >> 2, ch = idx & 3;          \
                async_ld_b128(vtp + (size_t)dr * SEQ + (k0) + 8 * ch,              \
                              Vts[buf] + dr * 80 + 16 * ch); }                     \
        } while (0)

    STAGE(0, 0);

    for (int t = 0; t <= t_max; ++t) {
        const int k0  = 32 * t;
        const int cur = t & 1;

        if (t < t_max) { STAGE(cur ^ 1, k0 + 32); WAIT_ASYNC(4); }
        else           { WAIT_ASYNC(0); }
        __syncthreads();

        // S = Q K^T : preload 4 B fragments, pin, then 4 WMMAs
        v8f s0 = V8ZERO, s1 = V8ZERO;
        {
            v16bf bk00 = load_fragB(Ks[cur] +  0 * 144 +  0, 144, lane);
            v16bf bk01 = load_fragB(Ks[cur] +  0 * 144 + 64, 144, lane);
            v16bf bk10 = load_fragB(Ks[cur] + 16 * 144 +  0, 144, lane);
            v16bf bk11 = load_fragB(Ks[cur] + 16 * 144 + 64, 144, lane);
            PIN5(bk00, bk01, bk10, bk11, qf0);
            s0 = wmma_bf16(qf0, bk00, s0);
            s0 = wmma_bf16(qf1, bk01, s0);
            s1 = wmma_bf16(qf0, bk10, s1);
            s1 = wmma_bf16(qf1, bk11, s1);
        }

        // causal mask + online softmax (rows live across 16-lane halves)
        #pragma unroll
        for (int r = 0; r < 8; ++r) {
            int qr = q0w + r + 8 * lh;
            if (k0 + col      > qr) s0[r] = NEG_INF;
            if (k0 + 16 + col > qr) s1[r] = NEG_INF;

            float loc = fmaxf(s0[r], s1[r]);
            loc = fmaxf(loc, __shfl_xor(loc, 8, 16));
            loc = fmaxf(loc, __shfl_xor(loc, 4, 16));
            loc = fmaxf(loc, __shfl_xor(loc, 2, 16));
            loc = fmaxf(loc, __shfl_xor(loc, 1, 16));

            float mnew  = fmaxf(m_run[r], loc);
            float alpha = __expf(m_run[r] - mnew);
            float p0 = __expf(s0[r] - mnew);
            float p1 = __expf(s1[r] - mnew);
            float sum = p0 + p1;
            sum += __shfl_xor(sum, 8, 16);
            sum += __shfl_xor(sum, 4, 16);
            sum += __shfl_xor(sum, 2, 16);
            sum += __shfl_xor(sum, 1, 16);

            l_run[r] = l_run[r] * alpha + sum;
            m_run[r] = mnew;
            o0[r] *= alpha; o1[r] *= alpha; o2[r] *= alpha; o3[r] *= alpha;

            int m = r + 8 * lh;
            *(unsigned short*)(Psw + m * 80 + col * 2)        = f32_to_bf16(p0);
            *(unsigned short*)(Psw + m * 80 + (16 + col) * 2) = f32_to_bf16(p1);
        }

        // same-wave LDS RAW (C-layout -> A-layout transpose)
        asm volatile("s_wait_dscnt 0x0" ::: "memory");
        v16bf pf = load_fragA(Psw, 80, lane);

        // O += P V : 4 B fragments, pin, then 4 WMMAs
        {
            v16bf bv0 = load_fragB(Vts[cur] +  0 * 80, 80, lane);
            v16bf bv1 = load_fragB(Vts[cur] + 16 * 80, 80, lane);
            v16bf bv2 = load_fragB(Vts[cur] + 32 * 80, 80, lane);
            v16bf bv3 = load_fragB(Vts[cur] + 48 * 80, 80, lane);
            PIN5(pf, bv0, bv1, bv2, bv3);
            o0 = wmma_bf16(pf, bv0, o0);
            o1 = wmma_bf16(pf, bv1, o1);
            o2 = wmma_bf16(pf, bv2, o2);
            o3 = wmma_bf16(pf, bv3, o3);
        }
        __syncthreads();
    }
    #undef STAGE

    // normalize and store attention output bf16 [B,L,D], D = h*64 + d
    #pragma unroll
    for (int r = 0; r < 8; ++r) {
        float inv = 1.0f / l_run[r];
        int qr = q0w + r + 8 * lh;
        unsigned short* orow = o_ws + ((size_t)b * SEQ + qr) * DMODEL + h * HDIM;
        orow[0  + col] = f32_to_bf16(o0[r] * inv);
        orow[16 + col] = f32_to_bf16(o1[r] * inv);
        orow[32 + col] = f32_to_bf16(o2[r] * inv);
        orow[48 + col] = f32_to_bf16(o3[r] * inv);
    }
}

// ---------------------------------------------------------------------------
// Kernel 3: output projection  out = attn_out @ W_o^T  (M=4096, N=1024, K=1024)
// A already bf16; register-staged double-buffered pipeline.
// ---------------------------------------------------------------------------
__global__ __launch_bounds__(256)
void oproj_gemm_wmma(const unsigned short* __restrict__ A,
                     const float* __restrict__ W,
                     float* __restrict__ out) {
    __shared__ alignas(16) unsigned char As[2][64 * 80];
    __shared__ alignas(16) unsigned char Bs[2][128 * 80];

    const int tid  = threadIdx.x;
    const int lane = tid & 31;
    const int wave = tid >> 5;
    const int lh   = lane >> 4;
    const int col  = lane & 15;

    const int mB0 = 64  * blockIdx.y;
    const int nB0 = 128 * blockIdx.x;
    const int mW  = 16 * (wave >> 1);
    const int nW  = 64 * (wave & 1);

    v8f acc0 = V8ZERO, acc1 = V8ZERO, acc2 = V8ZERO, acc3 = V8ZERO;

    uint4 areg; float4 br[4];
    const int arow = tid >> 2, ach = tid & 3;

    #define LOAD_A(k0)                                                         \
        areg = *(const uint4*)(A + (size_t)(mB0 + arow) * DMODEL + (k0) + 8 * ach);
    #define LOAD_B(k0)                                                         \
        _Pragma("unroll") for (int i = 0; i < 4; ++i) {                        \
            int idx = tid + 256 * i; int row = idx >> 3, c4 = idx & 7;         \
            br[i] = *(const float4*)(W + (size_t)(nB0 + row) * DMODEL + (k0) + 4 * c4); }
    #define STORE_A(buf) *(uint4*)((buf) + arow * 80 + 16 * ach) = areg;
    #define STORE_B(buf)                                                       \
        _Pragma("unroll") for (int i = 0; i < 4; ++i) {                        \
            int idx = tid + 256 * i; int row = idx >> 3, c4 = idx & 7;         \
            cvt4_store((buf) + row * 80 + 8 * c4, br[i]); }

    LOAD_A(0); LOAD_B(0);
    STORE_A(As[0]); STORE_B(Bs[0]);
    __syncthreads();

    for (int kt = 0; kt < DMODEL / 32; ++kt) {
        const int cur = kt & 1;
        const bool pre = (kt + 1 < DMODEL / 32);
        if (pre) { LOAD_A(32 * (kt + 1)); LOAD_B(32 * (kt + 1)); }

        v16bf a  = load_fragA(As[cur] + mW * 80, 80, lane);
        v16bf b0 = load_fragB(Bs[cur] + (nW +  0) * 80, 80, lane);
        v16bf b1 = load_fragB(Bs[cur] + (nW + 16) * 80, 80, lane);
        v16bf b2 = load_fragB(Bs[cur] + (nW + 32) * 80, 80, lane);
        v16bf b3 = load_fragB(Bs[cur] + (nW + 48) * 80, 80, lane);
        PIN5(a, b0, b1, b2, b3);
        acc0 = wmma_bf16(a, b0, acc0);
        acc1 = wmma_bf16(a, b1, acc1);
        acc2 = wmma_bf16(a, b2, acc2);
        acc3 = wmma_bf16(a, b3, acc3);

        if (pre) { STORE_A(As[cur ^ 1]); STORE_B(Bs[cur ^ 1]); }
        __syncthreads();
    }
    #undef LOAD_A
    #undef LOAD_B
    #undef STORE_A
    #undef STORE_B

    const int mB = mB0 + mW;
    const int nB = nB0 + nW;
    v8f accs[4] = { acc0, acc1, acc2, acc3 };
    #pragma unroll
    for (int nt = 0; nt < 4; ++nt) {
        #pragma unroll
        for (int r = 0; r < 8; ++r) {
            int m = mB + r + 8 * lh;
            int n = nB + 16 * nt + col;
            out[(size_t)m * DMODEL + n] = accs[nt][r];
        }
    }
}

// ---------------------------------------------------------------------------
// Launcher
// ---------------------------------------------------------------------------
extern "C" void kernel_launch(void* const* d_in, const int* in_sizes, int n_in,
                              void* d_out, int out_size, void* d_ws, size_t ws_size,
                              hipStream_t stream) {
    const float* x     = (const float*)d_in[0];   // [2,2048,1024]
    const float* W_qkv = (const float*)d_in[1];   // [3072,1024]
    const float* W_o   = (const float*)d_in[2];   // [1024,1024]
    float* out = (float*)d_out;

    const size_t PER = (size_t)BATCH * NHEAD * SEQ * HDIM;   // 4,194,304 elements
    unsigned short* q_ws  = (unsigned short*)d_ws;
    unsigned short* k_ws  = q_ws  + PER;
    unsigned short* vt_ws = k_ws  + PER;
    unsigned short* o_ws  = vt_ws + PER;

    qkv_gemm_wmma<<<dim3(NQKV / 128, MROWS / 64), 256, 0, stream>>>(
        x, W_qkv, q_ws, k_ws, vt_ws);

    flash_attn_wmma<<<dim3(SEQ / 64, NHEAD, BATCH), 128, 0, stream>>>(
        q_ws, k_ws, vt_ws, o_ws);

    oproj_gemm_wmma<<<dim3(DMODEL / 128, MROWS / 64), 256, 0, stream>>>(
        o_ws, W_o, out);
}
